// HFGA_54606214201918
// MI455X (gfx1250) — matrix operand order repaired and checked by reading, not compiled, not run on hardware
//
#include <hip/hip_runtime.h>
#include <hip/hip_bf16.h>

#define B_DIM 8
#define C_DIM 256
#define L_DIM 4096
#define M_DIM 2048

typedef _Float16 v16h __attribute__((ext_vector_type(16)));
typedef _Float16 h8   __attribute__((ext_vector_type(8)));
typedef float    v8f  __attribute__((ext_vector_type(8)));
typedef float    f4   __attribute__((ext_vector_type(4)));

__device__ __forceinline__ v16h pack2(h8 a, h8 b) {
  return __builtin_shufflevector(a, b, 0,1,2,3,4,5,6,7,8,9,10,11,12,13,14,15);
}

__device__ __forceinline__ v8f wmma_f16(v16h a, v16h b, v8f c) {
  return __builtin_amdgcn_wmma_f32_16x16x32_f16(false, a, false, b, (short)0, c, false, false);
}

// Async global->LDS copy of 16 bytes per enabled lane (gfx1250, ASYNCcnt).
// VDST operand = LDS byte address; generic shared pointers carry the LDS
// offset in their low 32 bits (flat LDS aperture mapping, ISA 10.2).
__device__ __forceinline__ void async_cp_b128(void* lds_dst, const void* gsrc) {
  asm volatile("global_load_async_to_lds_b128 %0, %1, off"
               :: "v"((unsigned)(size_t)lds_dst),
                  "v"((unsigned long long)(size_t)gsrc)
               : "memory");
}

__device__ __forceinline__ void wait_async0() {
  asm volatile("s_wait_asynccnt 0x0" ::: "memory");
}

// A-fragment (16x32, f16 source, row-major [row][stride]).
// Lane m(0-15): row row0+m, K = c0+0..15; lanes 16-31: K = c0+16..31.
__device__ __forceinline__ v16h load_a16(const _Float16* p, int stride, int row0, int c0, int lane) {
  int m = lane & 15, hf = lane >> 4;
  const _Float16* r = p + (size_t)(row0 + m) * stride + c0 + hf * 16;
  h8 u0 = *(const h8*)(r);
  h8 u1 = *(const h8*)(r + 8);
  return pack2(u0, u1);
}

// B-fragment (32x16) where B[k][n] = src[col0+n][c0+k] (f16 source).
// Lane n(0-15): col col0+n, K = c0+0..15; lanes 16-31: K = c0+16..31.
__device__ __forceinline__ v16h load_bT16(const _Float16* p, int stride, int col0, int c0, int lane) {
  int n = lane & 15, hf = lane >> 4;
  const _Float16* r = p + (size_t)(col0 + n) * stride + c0 + hf * 16;
  h8 u0 = *(const h8*)(r);
  h8 u1 = *(const h8*)(r + 8);
  return pack2(u0, u1);
}

// B-fragment from fp32 source with cvt: B[k][n] = src[col0+n][c0+k].
__device__ __forceinline__ v16h load_bT32(const float* p, int stride, int col0, int c0, int lane) {
  int n = lane & 15, hf = lane >> 4;
  const float* r = p + (size_t)(col0 + n) * stride + c0 + hf * 16;
  f4 a = *(const f4*)(r);
  f4 b = *(const f4*)(r + 4);
  f4 c = *(const f4*)(r + 8);
  f4 d = *(const f4*)(r + 12);
  v16h v;
#pragma unroll
  for (int j = 0; j < 4; ++j) {
    v[j]      = (_Float16)a[j];
    v[4 + j]  = (_Float16)b[j];
    v[8 + j]  = (_Float16)c[j];
    v[12 + j] = (_Float16)d[j];
  }
  return v;
}

// A-fragment from fp32 source with cvt (16x32 row-major).
__device__ __forceinline__ v16h load_a32(const float* p, int stride, int row0, int c0, int lane) {
  int m = lane & 15, hf = lane >> 4;
  const float* r = p + (size_t)(row0 + m) * stride + c0 + hf * 16;
  f4 a = *(const f4*)(r);
  f4 b = *(const f4*)(r + 4);
  f4 c = *(const f4*)(r + 8);
  f4 d = *(const f4*)(r + 12);
  v16h v;
#pragma unroll
  for (int j = 0; j < 4; ++j) {
    v[j]      = (_Float16)a[j];
    v[4 + j]  = (_Float16)b[j];
    v[8 + j]  = (_Float16)c[j];
    v[12 + j] = (_Float16)d[j];
  }
  return v;
}

// ---------------------------------------------------------------------------
// Kernel 0: Haar high band + transpose + f16 conversion.
//   xT [B][L][C], xhT [B][M][C]
// ---------------------------------------------------------------------------
__global__ void k_dwt_transpose(const float* __restrict__ x,
                                _Float16* __restrict__ xT,
                                _Float16* __restrict__ xhT) {
  const int b = blockIdx.z, c0 = blockIdx.y * 64, l0 = blockIdx.x * 64;
  __shared__ _Float16 sx[64 * 66];
  __shared__ _Float16 sxh[64 * 34];
  const int t = threadIdx.x;
  const float* xb = x + ((size_t)b * C_DIM + c0) * L_DIM + l0;
#pragma unroll
  for (int i = 0; i < 8; ++i) {
    int idx = i * 256 + t;
    int cc = idx >> 5, pr = idx & 31;
    float2 v = *(const float2*)(xb + (size_t)cc * L_DIM + pr * 2);
    sx[cc * 66 + pr * 2]     = (_Float16)v.x;
    sx[cc * 66 + pr * 2 + 1] = (_Float16)v.y;
    sxh[cc * 34 + pr] = (_Float16)((v.x - v.y) * 0.70710678118654752f);
  }
  __syncthreads();
  _Float16* xTb = xT + ((size_t)b * L_DIM + l0) * C_DIM + c0;
#pragma unroll
  for (int i = 0; i < 16; ++i) {
    int idx = i * 256 + t;
    int cc = idx & 63, ll = idx >> 6;
    xTb[(size_t)ll * C_DIM + cc] = sx[cc * 66 + ll];
  }
  _Float16* xhTb = xhT + ((size_t)b * M_DIM + (l0 >> 1)) * C_DIM + c0;
#pragma unroll
  for (int i = 0; i < 8; ++i) {
    int idx = i * 256 + t;
    int cc = idx & 63, mm = idx >> 6;
    xhTb[(size_t)mm * C_DIM + cc] = sxh[cc * 34 + mm];
  }
}

// ---------------------------------------------------------------------------
// Kernel 1a: token-major projection  D[row][o] = sum_c A[row][c]*W[o][c] + b[o]
//   Used for qT (R=L, A=xT) and kT (R=M, A=xhT). One wave per 16x16 tile.
// ---------------------------------------------------------------------------
__global__ void k_proj_rows(const _Float16* __restrict__ A, const float* __restrict__ W,
                            const float* __restrict__ bias, _Float16* __restrict__ D, int R) {
  const int lane = threadIdx.x & 31;
  const int tile = blockIdx.x * 4 + (threadIdx.x >> 5);
  const int tiles_per_b = (R >> 4) * 16;
  const int b = tile / tiles_per_b, rem = tile % tiles_per_b;
  const int rt = rem >> 4, ot = rem & 15;
  const _Float16* Ab = A + (size_t)b * R * C_DIM;
  v8f d = {};
#pragma unroll
  for (int cs = 0; cs < 8; ++cs) {
    v16h a = load_a16(Ab, C_DIM, rt * 16, cs * 32, lane);
    v16h w = load_bT32(W, C_DIM, ot * 16, cs * 32, lane);
    d = wmma_f16(a, w, d);
  }
  const int n = lane & 15, hf = lane >> 4;
  const float bb = bias[ot * 16 + n];
  _Float16* Db = D + (size_t)b * R * C_DIM;
#pragma unroll
  for (int r = 0; r < 8; ++r) {
    int row = rt * 16 + r + hf * 8;
    Db[(size_t)row * C_DIM + ot * 16 + n] = (_Float16)(d[r] + bb);
  }
}

// ---------------------------------------------------------------------------
// Kernel 1b: v[o][m] = sum_c Wv[o][c]*xh[c][m] + bv[o], stored key-blocked:
//   vB[b][m/32][c][m%32]  so attention's V A-fragments are contiguous.
// ---------------------------------------------------------------------------
__global__ void k_proj_v(const float* __restrict__ W, const _Float16* __restrict__ xhT,
                         const float* __restrict__ bias, _Float16* __restrict__ VB) {
  const int lane = threadIdx.x & 31;
  const int tile = blockIdx.x * 4 + (threadIdx.x >> 5);
  const int b = tile >> 11, rem = tile & 2047;  // 16 o-tiles * 128 m-tiles
  const int ot = rem >> 7, mt = rem & 127;
  const _Float16* xb = xhT + (size_t)b * M_DIM * C_DIM;
  v8f d = {};
#pragma unroll
  for (int cs = 0; cs < 8; ++cs) {
    v16h a = load_a32(W, C_DIM, ot * 16, cs * 32, lane);
    v16h bf = load_bT16(xb, C_DIM, mt * 16, cs * 32, lane);
    d = wmma_f16(a, bf, d);
  }
  const int n = lane & 15, hf = lane >> 4;
  const int m0 = mt * 16;
  _Float16* vb = VB + (size_t)b * C_DIM * M_DIM + (size_t)(m0 >> 5) * (C_DIM * 32) + (m0 & 31) + n;
#pragma unroll
  for (int r = 0; r < 8; ++r) {
    int c = ot * 16 + r + hf * 8;
    vb[(size_t)c * 32] = (_Float16)(d[r] + bias[c]);
  }
}

// ---------------------------------------------------------------------------
// Kernel 2: flash attention + gate + residual.
//   Block = 256 threads (8 waves), 128 queries per block (16/wave, full C).
//   Q staged once; K/V double-buffered via GLOBAL_LOAD_ASYNC_TO_LDS_B128.
// ---------------------------------------------------------------------------
#define LQS 264
#define LKS 264
#define LVS 40

// Issue async copies for one 32-key K block (32x512B) and V block (256x64B).
__device__ __forceinline__ void stage_kv_async(const _Float16* ksrc, const _Float16* vsrc,
                                               _Float16* lk, _Float16* lv, int t) {
#pragma unroll
  for (int i = 0; i < 4; ++i) {
    int idx = i * 256 + t;
    int row = idx >> 5, ch = idx & 31;
    async_cp_b128(lk + row * LKS + ch * 8, ksrc + (size_t)row * C_DIM + ch * 8);
  }
#pragma unroll
  for (int i = 0; i < 4; ++i) {
    int idx = i * 256 + t;
    int row = idx >> 2, ch = idx & 3;
    async_cp_b128(lv + row * LVS + ch * 8, vsrc + row * 32 + ch * 8);
  }
}

__global__ void __launch_bounds__(256)
k_attn(const _Float16* __restrict__ qT, const _Float16* __restrict__ kT,
       const _Float16* __restrict__ VB, const float* __restrict__ x,
       const float* __restrict__ gate, float* __restrict__ out) {
  extern __shared__ _Float16 smem[];
  _Float16* lq  = smem;                                  // 128*264 halfs
  _Float16* lk0 = smem + 128 * LQS;                      // 32*264
  _Float16* lk1 = lk0 + 32 * LKS;                        // 32*264
  _Float16* lv0 = lk1 + 32 * LKS;                        // 256*40
  _Float16* lv1 = lv0 + 256 * LVS;                       // 256*40
  _Float16* lkb[2] = {lk0, lk1};
  _Float16* lvb[2] = {lv0, lv1};

  const int t = threadIdx.x, lane = t & 31, wave = t >> 5;
  const int b = blockIdx.y, l0 = blockIdx.x * 128;

  // Stage 128 query rows of qT (contiguous 512B rows) into padded LDS (async).
  const _Float16* qsrc = qT + ((size_t)b * L_DIM + l0) * C_DIM;
#pragma unroll
  for (int i = 0; i < 16; ++i) {
    int idx = i * 256 + t;
    int row = idx >> 5, ch = idx & 31;
    async_cp_b128(lq + row * LQS + ch * 8, qsrc + (size_t)row * C_DIM + ch * 8);
  }

  const _Float16* kbase = kT + (size_t)b * M_DIM * C_DIM;
  const _Float16* vbase = VB + (size_t)b * C_DIM * M_DIM;
  // Prime the pipeline with key block 0.
  stage_kv_async(kbase, vbase, lkb[0], lvb[0], t);

  const int q0 = wave * 16;
  v8f acc[16];
#pragma unroll
  for (int i = 0; i < 16; ++i) acc[i] = (v8f){};
  float m_run = -__builtin_inff();
  float l_run = 0.f;
  const float sc = 0.0625f;  // 1/sqrt(C)
  int cur = 0;

  for (int kb = 0; kb < M_DIM; kb += 32) {
    wait_async0();      // my async copies for buffer `cur` (and lq) are done
    __syncthreads();    // everyone's are done; prev reads of buffer cur^1 done

    if (kb + 32 < M_DIM)
      stage_kv_async(kbase + (size_t)(kb + 32) * C_DIM,
                     vbase + (size_t)((kb + 32) >> 5) * (C_DIM * 32),
                     lkb[cur ^ 1], lvb[cur ^ 1], t);

    const _Float16* lk = lkb[cur];
    const _Float16* lv = lvb[cur];

    // S = K^T Q (fp32 accumulate): rows = keys, cols = queries.
    v8f S0 = {}, S1 = {};
#pragma unroll
    for (int cs = 0; cs < 8; ++cs) {
      int c0 = cs * 32;
      v16h bq = load_bT16(lq, LQS, q0, c0, lane);
      v16h a0 = load_a16(lk, LKS, 0, c0, lane);
      v16h a1 = load_a16(lk, LKS, 16, c0, lane);
      S0 = wmma_f16(a0, bq, S0);
      S1 = wmma_f16(a1, bq, S1);
    }

    // Online softmax over the 32 keys of this block (column = query).
    float mt = -__builtin_inff();
#pragma unroll
    for (int i = 0; i < 8; ++i) {
      S0[i] *= sc; S1[i] *= sc;
      mt = fmaxf(mt, fmaxf(S0[i], S1[i]));
    }
    mt = fmaxf(mt, __shfl_xor(mt, 16, 32));
    float mn = fmaxf(m_run, mt);
    float alpha = __expf(m_run - mn);
    float P0[8], P1[8];
    float ls = 0.f;
#pragma unroll
    for (int i = 0; i < 8; ++i) {
      P0[i] = __expf(S0[i] - mn);
      P1[i] = __expf(S1[i] - mn);
      ls += P0[i] + P1[i];
    }
    ls += __shfl_xor(ls, 16, 32);
    l_run = l_run * alpha + ls;
    m_run = mn;
#pragma unroll
    for (int ct = 0; ct < 16; ++ct) {
#pragma unroll
      for (int i = 0; i < 8; ++i) acc[ct][i] *= alpha;
    }

    // Repack P (two 16x16 f32 C-tiles) into the f16 B-fragment (K=32 keys).
    const bool lo = (lane < 16);
    v16h bp;
#pragma unroll
    for (int i = 0; i < 8; ++i) {
      float o0 = __shfl_xor(P0[i], 16, 32);
      float o1 = __shfl_xor(P1[i], 16, 32);
      bp[i]     = (_Float16)(lo ? P0[i] : o1);
      bp[8 + i] = (_Float16)(lo ? o0 : P1[i]);
    }

    // O += V * P across full C (16 row-tiles).
#pragma unroll
    for (int ct = 0; ct < 16; ++ct) {
      v16h va = load_a16(lv, LVS, ct * 16, 0, lane);
      acc[ct] = wmma_f16(va, bp, acc[ct]);
    }
    cur ^= 1;
  }

  // Epilogue: normalize, gate, residual. Stores coalesce into 64B sectors.
  const float invl = 1.f / l_run;
  const float g = tanhf(gate[0]);
  const int n = lane & 15, hf = lane >> 4;
  const int ql = l0 + q0 + n;
#pragma unroll
  for (int ct = 0; ct < 16; ++ct) {
#pragma unroll
    for (int r = 0; r < 8; ++r) {
      int c = ct * 16 + r + hf * 8;
      size_t idx = ((size_t)b * C_DIM + c) * L_DIM + ql;
      out[idx] = acc[ct][r] * invl * g + x[idx];
    }
  }
}

// ---------------------------------------------------------------------------
extern "C" void kernel_launch(void* const* d_in, const int* in_sizes, int n_in,
                              void* d_out, int out_size, void* d_ws, size_t ws_size,
                              hipStream_t stream) {
  const float* x    = (const float*)d_in[0];
  const float* Wq   = (const float*)d_in[1];
  const float* bq   = (const float*)d_in[2];
  const float* Wk   = (const float*)d_in[3];
  const float* bk   = (const float*)d_in[4];
  const float* Wv   = (const float*)d_in[5];
  const float* bv   = (const float*)d_in[6];
  const float* gate = (const float*)d_in[7];
  float* out = (float*)d_out;

  char* ws = (char*)d_ws;
  _Float16* xT  = (_Float16*)(ws);              // B*L*C*2 = 16 MiB
  _Float16* xhT = (_Float16*)(ws + 16777216);   // B*M*C*2 =  8 MiB
  _Float16* qT  = (_Float16*)(ws + 25165824);   // 16 MiB
  _Float16* kT  = (_Float16*)(ws + 41943040);   //  8 MiB
  _Float16* vB  = (_Float16*)(ws + 50331648);   //  8 MiB

  dim3 g0(L_DIM / 64, C_DIM / 64, B_DIM);
  k_dwt_transpose<<<g0, 256, 0, stream>>>(x, xT, xhT);

  // q: B*(L/16)*(C/16) = 32768 tiles, 4 waves/block
  k_proj_rows<<<8192, 128, 0, stream>>>(xT, Wq, bq, qT, L_DIM);
  // k: B*(M/16)*(C/16) = 16384 tiles
  k_proj_rows<<<4096, 128, 0, stream>>>(xhT, Wk, bk, kT, M_DIM);
  // v: B*16*(M/16) = 16384 tiles
  k_proj_v<<<4096, 128, 0, stream>>>(Wv, xhT, bv, vB);

  dim3 g2(L_DIM / 128, B_DIM);
  size_t smem = (size_t)(128 * LQS + 2 * 32 * LKS + 2 * 256 * LVS) * sizeof(_Float16); // 142336 B
  k_attn<<<g2, 256, smem, stream>>>(qT, kT, vB, x, gate, out);
}
